// SpikeFFN_59399397704347
// MI455X (gfx1250) — matrix operand order repaired and checked by reading, not compiled, not run on hardware
//
#include <hip/hip_runtime.h>

// SpikeFFN fused kernel for gfx1250 (MI455X):
//   h = x @ W1^T -> multi-step LIF (tau=2, v_th=1, hard reset 0) -> out = s @ W2^T
// x [4, 8192, 512] f32, W1/W2 [512,512] f32, out [4, 8192, 512] f32.
//
// Per 256-thread block (8 waves): 16 rows of (B*N) x all 512 channels x all 4 T.
//  - TDM (tensor_load_to_lds) stages the 4x16x512 f32 x-tile into LDS
//  - one cooperative f32->f16 conversion pass into an f16 LDS tile
//  - stage 1: WMMA f16 GEMM (A from LDS, B = pre-converted f16 W1 from d_ws)
//  - LIF in registers, spikes (exact in f16) -> LDS (region reuses raw x tile)
//  - stage 2: WMMA f16 GEMM (A = spikes from LDS, B = f16 W2), f32 stores.

typedef __attribute__((ext_vector_type(16))) _Float16 v16h;
typedef __attribute__((ext_vector_type(8)))  _Float16 v8h;
typedef __attribute__((ext_vector_type(4)))  _Float16 v4h;
typedef __attribute__((ext_vector_type(8)))  float    v8f;
typedef __attribute__((ext_vector_type(4)))  float    v4f;
typedef unsigned int v4ui __attribute__((ext_vector_type(4)));
typedef int          v8si __attribute__((ext_vector_type(8)));
typedef int          v4si __attribute__((ext_vector_type(4)));

#define TSTEPS 4
#define ROWS   8192    // B * N
#define CIN    512
#define COUT   512

// LDS layout (dynamic, 192 KB):
//   [0      , 128K) raw f32 x tile [4][16][512]  (dead after conversion pass)
//   [0      ,  64K) spike tile [4][16][512] f16  (reuses raw region later)
//   [128K   , 192K) f16 x tile [4][16][512]
#define LDS_X16_OFF  131072
#define LDS_BYTES    196608

__device__ __forceinline__ v8f vzero8() { v8f z = {}; return z; }

// A-fragment (16x32 f16) from an f16 tile, row-major [ld]: per-lane two
// contiguous 8-half runs at k0 + 8*(lane>=16) and +16 (ISA 7.12.2).
__device__ __forceinline__ v16h frag_a_h16(const _Float16* p) {
  v8h lo = *(const v8h*)(p);
  v8h hi = *(const v8h*)(p + 16);
  v16h a;
#pragma unroll
  for (int i = 0; i < 8; ++i) { a[i] = lo[i]; a[8 + i] = hi[i]; }
  return a;
}

// B-fragment (32x16 f16) for D=A*B with B[k][n] = W[n][k], W f16 row-major [ld]:
// per-lane two contiguous 8-half runs at k0 + 8*(lane>=16) and +16.
__device__ __forceinline__ v16h frag_b_h16(const _Float16* __restrict__ W, int ld,
                                           int n0, int k0, int lane) {
  const _Float16* p = W + (size_t)(n0 + (lane & 15)) * ld + k0 + ((lane >> 4) * 8);
  return frag_a_h16(p);   // 16B aligned -> two b128 loads
}

// ---- weight f32 -> f16 pre-conversion (one float4 per thread) ----------------
__global__ __launch_bounds__(256)
void cvt_w_kernel(const float* __restrict__ src, _Float16* __restrict__ dst) {
  const int i = blockIdx.x * 256 + threadIdx.x;   // i < 512*512/4
  v4f f = ((const v4f*)src)[i];
  v4h h;
#pragma unroll
  for (int j = 0; j < 4; ++j) h[j] = (_Float16)f[j];
  ((v4h*)dst)[i] = h;
}

__global__ __launch_bounds__(256)
void spikeffn_fused_kernel(const float* __restrict__ x,
                           const _Float16* __restrict__ W1h,
                           const _Float16* __restrict__ W2h,
                           float* __restrict__ out) {
  extern __shared__ char smem[];
  float*    Xlds32 = (float*)smem;
  _Float16* Xlds16 = (_Float16*)(smem + LDS_X16_OFF);
  _Float16* Slds   = (_Float16*)smem;

  const int tid   = threadIdx.x;
  const int wave  = tid >> 5;
  const int lane  = tid & 31;
  const int hh    = lane >> 4;
  const int ncol  = lane & 15;
  const int m0    = blockIdx.x * 16;
  const int cbase = wave * 64;

  // ---- Stage x tile into LDS (raw f32) via the Tensor Data Mover -------------
#if __has_builtin(__builtin_amdgcn_tensor_load_to_lds)
  if (wave == 0) {
#pragma unroll 1
    for (int t = 0; t < TSTEPS; ++t) {
      const unsigned lds_addr =
          (unsigned)(size_t)(smem + (size_t)t * 16 * CIN * sizeof(float));
      const unsigned long long ga =
          (unsigned long long)(const void*)x +
          (((unsigned long long)t * ROWS + (unsigned long long)m0) * CIN) * sizeof(float);
      // D# group 0: count=1 (valid), lds_addr, global_addr[56:0], type=2
      v4ui g0 = { 1u, lds_addr,
                  (unsigned)(ga & 0xFFFFFFFFu),
                  ((unsigned)((ga >> 32) & 0x01FFFFFFu)) | 0x80000000u };
      // D# group 1: data_size=4B; tensor_dim0=512; tensor_dim1=8192;
      //             tile_dim0=512, tile_dim1=16; tensor_dim0_stride=512
      v8si g1 = { (int)0x00020000,
                  (int)(512u  << 16),
                  (int)(8192u << 16),
                  (int)(512u  << 16),
                  16, 512, 0, 0 };
      v4si gz = { 0, 0, 0, 0 };
#if __clang_major__ >= 23
      v8si gz8 = { 0, 0, 0, 0, 0, 0, 0, 0 };
      __builtin_amdgcn_tensor_load_to_lds(g0, g1, gz, gz, gz8, 0);
#else
      __builtin_amdgcn_tensor_load_to_lds(g0, g1, gz, gz, 0);
#endif
    }
    __builtin_amdgcn_s_wait_tensorcnt(0);
  }
#else
  // Fallback: cooperative coalesced staging
  for (int c = tid; c < TSTEPS * 16 * CIN / 4; c += 256) {
    const int t   = c / (16 * CIN / 4);
    const int rem = c % (16 * CIN / 4);
    const int row = rem / (CIN / 4);
    const int kk  = rem % (CIN / 4);
    ((v4f*)Xlds32)[c] =
        ((const v4f*)(x + ((size_t)t * ROWS + m0 + row) * CIN))[kk];
  }
#endif
  __syncthreads();

  // ---- One-shot f32 -> f16 conversion of the x tile --------------------------
  for (int c = tid; c < TSTEPS * 16 * CIN / 8; c += 256) {
    const float* s = Xlds32 + (size_t)c * 8;
    v4f f0 = *(const v4f*)(s);
    v4f f1 = *(const v4f*)(s + 4);
    v8h h;
#pragma unroll
    for (int i = 0; i < 4; ++i) {
      h[i]     = (_Float16)f0[i];
      h[4 + i] = (_Float16)f1[i];
    }
    *(v8h*)(Xlds16 + (size_t)c * 8) = h;
  }
  __syncthreads();   // Xlds32 dead beyond this point; region reused for spikes

  // ---- Stage 1: H = X @ W1^T, all 4 timesteps --------------------------------
  v8f acc[4][TSTEPS];
#pragma unroll
  for (int ct = 0; ct < 4; ++ct)
#pragma unroll
    for (int t = 0; t < TSTEPS; ++t) acc[ct][t] = vzero8();

  for (int k0 = 0; k0 < CIN; k0 += 32) {
    v16h afr[TSTEPS];
#pragma unroll
    for (int t = 0; t < TSTEPS; ++t)
      afr[t] = frag_a_h16(Xlds16 + ((t * 16 + ncol) * CIN) + k0 + hh * 8);
#pragma unroll
    for (int ct = 0; ct < 4; ++ct) {
      v16h bfr = frag_b_h16(W1h, CIN, cbase + ct * 16, k0, lane);
#pragma unroll
      for (int t = 0; t < TSTEPS; ++t)
        acc[ct][t] = __builtin_amdgcn_wmma_f32_16x16x32_f16(
            false, afr[t], false, bfr, (short)0, acc[ct][t], false, false);
    }
  }

  // ---- LIF (tau=2, v_th=1, hard reset to 0) -> spikes in LDS -----------------
#pragma unroll
  for (int ct = 0; ct < 4; ++ct) {
    const int c0 = cbase + ct * 16;
    v8f v = vzero8();
#pragma unroll
    for (int t = 0; t < TSTEPS; ++t) {
#pragma unroll
      for (int i = 0; i < 8; ++i) {
        float vn = 0.5f * (v[i] + acc[ct][t][i]);   // v + (h - v)/tau
        bool  sp = (vn >= 1.0f);
        v[i] = sp ? 0.0f : vn;                      // hard reset
        const int r = i + hh * 8;                   // C/D layout row
        Slds[((t * 16 + r) * COUT) + c0 + ncol] = (_Float16)(sp ? 1.0f : 0.0f);
      }
    }
  }
  __syncthreads();

  // ---- Stage 2: out = S @ W2^T -----------------------------------------------
#pragma unroll 1
  for (int t = 0; t < TSTEPS; ++t) {
    v8f oacc[4];
#pragma unroll
    for (int ct = 0; ct < 4; ++ct) oacc[ct] = vzero8();

    const _Float16* st = &Slds[t * 16 * COUT];
    for (int k0 = 0; k0 < COUT; k0 += 32) {
      v16h afr = frag_a_h16(st + (ncol * COUT) + k0 + hh * 8);
#pragma unroll
      for (int ct = 0; ct < 4; ++ct) {
        v16h bfr = frag_b_h16(W2h, COUT, cbase + ct * 16, k0, lane);
        oacc[ct] = __builtin_amdgcn_wmma_f32_16x16x32_f16(
            false, afr, false, bfr, (short)0, oacc[ct], false, false);
      }
    }

#pragma unroll
    for (int ct = 0; ct < 4; ++ct) {
      const int p0 = cbase + ct * 16;
#pragma unroll
      for (int i = 0; i < 8; ++i) {
        const int r = i + hh * 8;
        out[((size_t)t * ROWS + m0 + r) * COUT + p0 + ncol] = oacc[ct][i];
      }
    }
  }
}

extern "C" void kernel_launch(void* const* d_in, const int* in_sizes, int n_in,
                              void* d_out, int out_size, void* d_ws, size_t ws_size,
                              hipStream_t stream) {
  (void)in_sizes; (void)n_in; (void)out_size; (void)ws_size;
  const float* x  = (const float*)d_in[0];
  const float* W1 = (const float*)d_in[1];
  const float* W2 = (const float*)d_in[2];
  float* out = (float*)d_out;

  _Float16* W1h = (_Float16*)d_ws;            // 512 KB
  _Float16* W2h = W1h + 512 * 512;            // 512 KB

  cvt_w_kernel<<<dim3(256), dim3(256), 0, stream>>>(W1, W1h);
  cvt_w_kernel<<<dim3(256), dim3(256), 0, stream>>>(W2, W2h);

  (void)hipFuncSetAttribute((const void*)spikeffn_fused_kernel,
                            hipFuncAttributeMaxDynamicSharedMemorySize, LDS_BYTES);
  spikeffn_fused_kernel<<<dim3(ROWS / 16), dim3(256), LDS_BYTES, stream>>>(
      x, W1h, W2h, out);
}
